// Attension_Point_55929064129268
// MI455X (gfx1250) — matrix operand order repaired and checked by reading, not compiled, not run on hardware
//
#include <hip/hip_runtime.h>
#include <math.h>

// ---------------------------------------------------------------------------
// CDNA5 (gfx1250) Attention-Point network. Heavy math on v_wmma_f32_16x16x32_f16.
// Activations live in LDS in fragment-major layout (40-half / 80B padded records
// -> 16B-aligned ds_load_b128 / ds_store_b128, bank-swizzled). Weights are
// pre-packed into A-fragment order so each A fragment is 2 global b128 loads.
// ---------------------------------------------------------------------------

typedef __attribute__((ext_vector_type(16))) _Float16 v16h;
typedef __attribute__((ext_vector_type(8)))  _Float16 v8h;
typedef __attribute__((ext_vector_type(8)))  float    v8f;

// 16 contiguous halves (32B, 16B-aligned) -> v16h via two b128 loads.
__device__ __forceinline__ v16h frag_ld16(const _Float16* p) {
  v8h lo = *(const v8h*)p;
  v8h hi = *(const v8h*)(p + 8);
  return __builtin_shufflevector(lo, hi, 0, 1, 2, 3, 4, 5, 6, 7,
                                 8, 9, 10, 11, 12, 13, 14, 15);
}

// A fragments pre-packed by k_prep_w: frag (mt,kt) = 32 lanes x 16 halves.
__device__ __forceinline__ v16h frag_a(const _Float16* __restrict__ Wf,
                                       int ktiles, int mt, int kt, int lane) {
  return frag_ld16(Wf + ((size_t)((mt * ktiles + kt) << 5) + lane) * 16);
}

// Activation LDS layout: plane kt (32 channels) -> [point][40 halves]
// (halves 0..31 = k 0..31, 8 pad). B fragment: lanes 0-15 read k 0..15 of
// point n0+(lane&15); lanes 16-31 read k 16..31.
__device__ __forceinline__ v16h frag_b(const _Float16* Xf, int npts,
                                       int kt, int n0, int lane) {
  return frag_ld16(Xf + (size_t)(kt * npts + n0 + (lane & 15)) * 40 +
                   ((lane >> 4) << 4));
}

// One conv layer: Xout = relu(W*Xin + bias). Bias pre-seeded into the WMMA
// accumulator (C rows = m channels broadcast over n). A reused across NT tiles.
template <int KT, int MT, int NT>
__device__ __forceinline__ void wmma_layer(
    const _Float16* __restrict__ Wf, const float* __restrict__ bias,
    const _Float16* Xin, _Float16* Xout, int lane, int wave) {
  constexpr int NP = NT << 4;
  for (int mt = wave; mt < MT; mt += 4) {
    const int mb = (mt << 4) + ((lane >> 4) << 3);
    float b8[8];
#pragma unroll
    for (int v = 0; v < 8; ++v) b8[v] = bias[mb + v];
    v8f acc[NT];
#pragma unroll
    for (int nt = 0; nt < NT; ++nt)
#pragma unroll
      for (int v = 0; v < 8; ++v) acc[nt][v] = b8[v];
#pragma unroll 4
    for (int kt = 0; kt < KT; ++kt) {
      v16h a = frag_a(Wf, KT, mt, kt, lane);
#pragma unroll
      for (int nt = 0; nt < NT; ++nt) {
        v16h b = frag_b(Xin, NP, kt, nt << 4, lane);
        acc[nt] = __builtin_amdgcn_wmma_f32_16x16x32_f16(
            false, a, false, b, (short)0, acc[nt], false, false);
      }
    }
#pragma unroll
    for (int nt = 0; nt < NT; ++nt) {
      const int n = (nt << 4) + (lane & 15);
      v8h h;
#pragma unroll
      for (int v = 0; v < 8; ++v) h[v] = (_Float16)fmaxf(acc[nt][v], 0.f);
      *(v8h*)(Xout + (size_t)((mb >> 5) * NP + n) * 40 + (mb & 31)) = h;
    }
  }
}

// Final layer of a tower: relu + max-pool over all NP points fused in regs
// (wave32 shfl_xor butterfly over the 16 lanes of each C-fragment half).
template <int KT, int MT, int NT>
__device__ __forceinline__ void wmma_layer_max(
    const _Float16* __restrict__ Wf, const float* __restrict__ bias,
    const _Float16* Xin, float* __restrict__ outBase, int outStride,
    int lane, int wave) {
  constexpr int NP = NT << 4;
  for (int mt = wave; mt < MT; mt += 4) {
    const int mb = (mt << 4) + ((lane >> 4) << 3);
    float b8[8];
#pragma unroll
    for (int v = 0; v < 8; ++v) b8[v] = bias[mb + v];
    v8f acc[NT];
#pragma unroll
    for (int nt = 0; nt < NT; ++nt)
#pragma unroll
      for (int v = 0; v < 8; ++v) acc[nt][v] = b8[v];
#pragma unroll 4
    for (int kt = 0; kt < KT; ++kt) {
      v16h a = frag_a(Wf, KT, mt, kt, lane);
#pragma unroll
      for (int nt = 0; nt < NT; ++nt) {
        v16h b = frag_b(Xin, NP, kt, nt << 4, lane);
        acc[nt] = __builtin_amdgcn_wmma_f32_16x16x32_f16(
            false, a, false, b, (short)0, acc[nt], false, false);
      }
    }
    float mx[8];
#pragma unroll
    for (int v = 0; v < 8; ++v) mx[v] = 0.f;  // relu folded into the max
#pragma unroll
    for (int nt = 0; nt < NT; ++nt)
#pragma unroll
      for (int v = 0; v < 8; ++v) mx[v] = fmaxf(mx[v], acc[nt][v]);
#pragma unroll
    for (int off = 1; off < 16; off <<= 1)
#pragma unroll
      for (int v = 0; v < 8; ++v)
        mx[v] = fmaxf(mx[v], __shfl_xor(mx[v], off, 32));
    if ((lane & 15) == 0)
#pragma unroll
      for (int v = 0; v < 8; ++v)
        outBase[(size_t)(mb + v) * outStride] = mx[v];
  }
}

// ------------------------- weight preparation ------------------------------
// Pack W (cout x cin, f32) into f16 A-fragment order, gamma folded, K padded
// to cinP. bOut = b*g + be.
__global__ void k_prep_w(const float* __restrict__ W, const float* __restrict__ b,
                         const float* __restrict__ g, const float* __restrict__ be,
                         _Float16* __restrict__ Wout, float* __restrict__ bOut,
                         int cin, int cinP, int cout) {
  int i = blockIdx.x * blockDim.x + threadIdx.x;
  int total = cout * cinP;
  if (i < total) {
    int ktiles = cinP >> 5;
    int fi = i >> 9, r = i & 511;
    int lane = r >> 4, h = r & 15;
    int mt = fi / ktiles, kt = fi - mt * ktiles;
    int m = (mt << 4) + (lane & 15);
    int kb = (lane >> 4) << 4;
    int k = (kt << 5) + kb + h;
    float v = (k < cin) ? W[(size_t)m * cin + k] * g[m] : 0.f;
    Wout[i] = (_Float16)v;
  }
  if (i < cout) bOut[i] = b[i] * g[i] + be[i];
}

// ------------------------- tower 1 (fused, + max over K) -------------------
// grid = B*N1 (16384) blocks, 128 threads. One (b, n) per block.
__global__ __launch_bounds__(128) void k_tower1(
    const float* __restrict__ xt, const float* __restrict__ yt,
    const _Float16* __restrict__ W1, const _Float16* __restrict__ W2,
    const _Float16* __restrict__ W3,
    const float* __restrict__ b1, const float* __restrict__ b2,
    const float* __restrict__ b3, float* __restrict__ x1out) {
  __shared__ alignas(32) _Float16 P0[2 * 64 * 40];  // Xin(1 kt) / X2(2 kt)
  __shared__ alignas(32) _Float16 P1[2 * 64 * 40];  // X1(2 kt)
  const int blk = blockIdx.x, b = blk >> 9, n = blk & 511;
  const int t = threadIdx.x, lane = t & 31, wave = t >> 5;

  for (int i = t; i < 32 * 64; i += 128) {
    int c = i >> 6, k = i & 63;
    float v = (c < 6) ? xt[(((size_t)b * 6 + c) * 512 + n) * 64 + k] : 0.f;
    P0[k * 40 + c] = (_Float16)v;
  }
  __syncthreads();
  wmma_layer<1, 4, 4>(W1, b1, P0, P1, lane, wave);
  __syncthreads();
  wmma_layer<2, 4, 4>(W2, b2, P1, P0, lane, wave);
  __syncthreads();
  wmma_layer_max<2, 8, 4>(W3, b3, P0,
                          x1out + ((size_t)b * 131 + 3) * 512 + n, 512,
                          lane, wave);
  if (t < 3) x1out[((size_t)b * 131 + t) * 512 + n] = yt[((size_t)b * 3 + t) * 512 + n];
}

// ------------- tower 2: fused kNN selection + gather + conv + max ----------
// grid = B*N2 (4096) blocks, 128 threads. One (b, center) per block.
__global__ __launch_bounds__(128) void k_tower2(
    const float* __restrict__ x1,
    const _Float16* __restrict__ W1, const _Float16* __restrict__ W2,
    const _Float16* __restrict__ W3,
    const float* __restrict__ b1, const float* __restrict__ b2,
    const float* __restrict__ b3, float* __restrict__ x2out) {
  __shared__ alignas(32) _Float16 P0[5 * 64 * 40];  // Xin(5 kt) / X2(4 kt)
  __shared__ alignas(32) _Float16 P1[4 * 64 * 40];  // X1(4 kt)
  __shared__ float d2[512];
  __shared__ float rv[128];
  __shared__ int   ri[128];
  __shared__ int   sel[64];
  __shared__ float ctr[3];
  const int blk = blockIdx.x, b = blk >> 7, c0 = blk & 127;
  const int t = threadIdx.x, lane = t & 31, wave = t >> 5;
  const float* xb = x1 + (size_t)b * 131 * 512;

  if (t < 3) ctr[t] = xb[t * 512 + c0];
  __syncthreads();
  for (int j = t; j < 512; j += 128) {
    float dx = xb[0 * 512 + j] - ctr[0];
    float dy = xb[1 * 512 + j] - ctr[1];
    float dz = xb[2 * 512 + j] - ctr[2];
    d2[j] = dx * dx + dy * dy + dz * dz;
  }
  __syncthreads();
  // 64 nearest (set semantics; downstream only max-pools over k).
  for (int s = 0; s < 64; ++s) {
    float best = 1e30f; int bi = 0x7fffffff;
    for (int j = t; j < 512; j += 128) {
      float v = d2[j];
      if (v < best || (v == best && j < bi)) { best = v; bi = j; }
    }
    rv[t] = best; ri[t] = bi;
    __syncthreads();
    for (int off = 64; off > 0; off >>= 1) {
      if (t < off) {
        float v = rv[t + off]; int j = ri[t + off];
        if (v < rv[t] || (v == rv[t] && j < ri[t])) { rv[t] = v; ri[t] = j; }
      }
      __syncthreads();
    }
    if (t == 0) {
      int j = ri[0];
      sel[s] = (rv[0] > 0.5f) ? c0 : j;  // RADIUS2 -> replace with center idx
      d2[j] = 1e30f;
    }
    __syncthreads();
  }
  // Gather 131 channels x 64 neighbors (center-relative xyz), pad to 160.
  for (int i = t; i < 160 * 64; i += 128) {
    int c = i >> 6, k = i & 63;
    float v = 0.f;
    if (c < 131) {
      v = xb[c * 512 + sel[k]];
      if (c < 3) v -= ctr[c];
    }
    P0[(size_t)((c >> 5) * 64 + k) * 40 + (c & 31)] = (_Float16)v;
  }
  __syncthreads();
  wmma_layer<5, 8, 4>(W1, b1, P0, P1, lane, wave);
  __syncthreads();
  wmma_layer<4, 8, 4>(W2, b2, P1, P0, lane, wave);
  __syncthreads();
  wmma_layer_max<4, 16, 4>(W3, b3, P0,
                           x2out + ((size_t)b * 259 + 3) * 128 + c0, 128,
                           lane, wave);
  if (t < 3) x2out[((size_t)b * 259 + t) * 128 + c0] = ctr[t];
}

// ----------------------------- non-local block -----------------------------
__global__ void k_nl_proj(const float* __restrict__ x2,
                          const float* __restrict__ Wg,  const float* __restrict__ bg,
                          const float* __restrict__ Wth, const float* __restrict__ bth,
                          const float* __restrict__ Wph, const float* __restrict__ bph,
                          float* __restrict__ gx, float* __restrict__ th,
                          float* __restrict__ ph) {
  const int per = 32 * 128 * 129;
  int i = blockIdx.x * blockDim.x + threadIdx.x;
  if (i >= 3 * per) return;
  int which = i / per, r = i - which * per;
  int b = r / (128 * 129), r2 = r - b * (128 * 129);
  int n = r2 / 129, c = r2 - n * 129;
  const float* W  = (which == 0) ? Wg : (which == 1) ? Wth : Wph;
  const float* bb = (which == 0) ? bg : (which == 1) ? bth : bph;
  const float* xf = x2 + (size_t)b * 259 * 128 + n;
  const float* w  = W + (size_t)c * 259;
  float s = 0.f;
  for (int k = 0; k < 259; ++k) s += w[k] * xf[(size_t)k * 128];
  s += bb[c];
  if (which == 0)      gx[((size_t)b * 128 + n) * 129 + c] = s;
  else if (which == 1) th[((size_t)b * 128 + n) * 129 + c] = s;
  else                 ph[((size_t)b * 129 + c) * 128 + n] = s;
}

__global__ __launch_bounds__(128) void k_nl_attn(const float* __restrict__ th,
                                                 const float* __restrict__ ph,
                                                 float* __restrict__ a) {
  __shared__ float sth[129];
  __shared__ float red[128];
  const int b = blockIdx.x >> 7, n = blockIdx.x & 127, t = threadIdx.x;
  for (int i = t; i < 129; i += 128) sth[i] = th[((size_t)b * 128 + n) * 129 + i];
  __syncthreads();
  const float* pb = ph + (size_t)b * 129 * 128;
  float f = 0.f;
  for (int i = 0; i < 129; ++i) f += sth[i] * pb[(size_t)i * 128 + t];
  red[t] = f; __syncthreads();
  for (int off = 64; off > 0; off >>= 1) {
    if (t < off) red[t] = fmaxf(red[t], red[t + off]);
    __syncthreads();
  }
  float mx = red[0]; __syncthreads();
  float e = __expf(f - mx);
  red[t] = e; __syncthreads();
  for (int off = 64; off > 0; off >>= 1) {
    if (t < off) red[t] += red[t + off];
    __syncthreads();
  }
  a[((size_t)b * 128 + n) * 128 + t] = e / red[0];
}

__global__ void k_nl_y(const float* __restrict__ a, const float* __restrict__ gx,
                       float* __restrict__ y) {
  int i = blockIdx.x * blockDim.x + threadIdx.x;
  if (i >= 32 * 128 * 129) return;
  int b = i / (128 * 129), r = i - b * (128 * 129);
  int n = r / 129, c = r - n * 129;
  const float* ar = a + ((size_t)b * 128 + n) * 128;
  const float* gc = gx + (size_t)b * 128 * 129 + c;
  float s = 0.f;
  for (int m = 0; m < 128; ++m) s += ar[m] * gc[(size_t)m * 129];
  y[i] = s;
}

__global__ void k_nl_out(const float* __restrict__ y, const float* __restrict__ Ww,
                         const float* __restrict__ bw, float* __restrict__ x2) {
  int i = blockIdx.x * blockDim.x + threadIdx.x;
  if (i >= 32 * 259 * 128) return;
  int b = i / (259 * 128), r = i - b * (259 * 128);
  int c = r / 128, n = r - c * 128;
  const float* yr = y + ((size_t)b * 128 + n) * 129;
  const float* w  = Ww + (size_t)c * 129;
  float s = bw[c];
  for (int k = 0; k < 129; ++k) s += w[k] * yr[k];
  x2[i] += s;
}

// ---------------- tower 3 (fused, + register max over points) -------------
// grid = B*4 (128) blocks, 128 threads; 32 points per block.
__global__ __launch_bounds__(128) void k_tower3(
    const float* __restrict__ x2,
    const _Float16* __restrict__ W1, const _Float16* __restrict__ W2,
    const _Float16* __restrict__ W3,
    const float* __restrict__ b1, const float* __restrict__ b2,
    const float* __restrict__ b3, float* __restrict__ part) {
  __shared__ alignas(32) _Float16 P0[16 * 32 * 40];  // Xin(9 kt) / X2(16 kt)
  __shared__ alignas(32) _Float16 P1[8 * 32 * 40];   // X1(8 kt)
  const int blk = blockIdx.x, b = blk >> 2, ch = blk & 3;
  const int t = threadIdx.x, lane = t & 31, wave = t >> 5;
  const int nbase = ch * 32;
  const float* xb = x2 + (size_t)b * 259 * 128;

  for (int i = t; i < 288 * 32; i += 128) {
    int c = i >> 5, k = i & 31;
    float v = (c < 259) ? xb[(size_t)c * 128 + nbase + k] : 0.f;
    P0[(size_t)((c >> 5) * 32 + k) * 40 + (c & 31)] = (_Float16)v;
  }
  __syncthreads();
  wmma_layer<9, 16, 2>(W1, b1, P0, P1, lane, wave);
  __syncthreads();
  wmma_layer<8, 32, 2>(W2, b2, P1, P0, lane, wave);
  __syncthreads();
  wmma_layer_max<16, 64, 2>(W3, b3, P0,
                            part + ((size_t)b * 4 + ch) * 1024, 1, lane, wave);
}

// ----------------------------- tail ----------------------------------------
__global__ void k_final_max_bn(const float* __restrict__ part,
                               const float* __restrict__ g,
                               const float* __restrict__ be,
                               float* __restrict__ feat) {
  int i = blockIdx.x * blockDim.x + threadIdx.x;
  if (i >= 32 * 1024) return;
  int b = i >> 10, c = i & 1023;
  float m = part[((size_t)b * 4 + 0) * 1024 + c];
  for (int j = 1; j < 4; ++j) m = fmaxf(m, part[((size_t)b * 4 + j) * 1024 + c]);
  feat[i] = fmaxf(m * g[c] + be[c], 0.f);
}

__global__ void k_fc1(const float* __restrict__ feat, const float* __restrict__ W,
                      const float* __restrict__ bias, const float* __restrict__ g,
                      const float* __restrict__ be, float* __restrict__ h) {
  int i = blockIdx.x * blockDim.x + threadIdx.x;
  if (i >= 32 * 256) return;
  int b = i >> 8, o = i & 255;
  const float* x = feat + (size_t)b * 1024;
  const float* w = W + (size_t)o * 1024;
  float s = 0.f;
  for (int k = 0; k < 1024; ++k) s += x[k] * w[k];
  h[i] = fmaxf((s + bias[o]) * g[o] + be[o], 0.f);
}

__global__ void k_fc2(const float* __restrict__ h, const float* __restrict__ W,
                      const float* __restrict__ bias, float* __restrict__ out) {
  int i = blockIdx.x * blockDim.x + threadIdx.x;
  if (i >= 32 * 60) return;
  int b = i / 60, o = i - b * 60;
  const float* x = h + (size_t)b * 256;
  const float* w = W + (size_t)o * 256;
  float s = 0.f;
  for (int k = 0; k < 256; ++k) s += x[k] * w[k];
  out[i] = s + bias[o];
}

// ---------------------------------------------------------------------------
extern "C" void kernel_launch(void* const* d_in, const int* in_sizes, int n_in,
                              void* d_out, int out_size, void* d_ws, size_t ws_size,
                              hipStream_t stream) {
  (void)in_sizes; (void)n_in; (void)out_size; (void)ws_size;
  const float* xt = (const float*)d_in[0];
  const float* yt = (const float*)d_in[1];
  const int iR1 = 2, iR2 = 14, iNL = 26, iR3 = 34;
  auto F = [&](int i) { return (const float*)d_in[i]; };

  // ---- workspace carve-out ----
  char* wsb = (char*)d_ws;
  size_t off = 0;
  auto alloc = [&](size_t bytes) -> void* {
    void* p = wsb + off;
    off += (bytes + 255) & ~(size_t)255;
    return p;
  };
  float* x1   = (float*)alloc((size_t)32 * 131 * 512 * 4);
  float* x2   = (float*)alloc((size_t)32 * 259 * 128 * 4);
  float* gx   = (float*)alloc((size_t)32 * 128 * 129 * 4);
  float* thb  = (float*)alloc((size_t)32 * 128 * 129 * 4);
  float* phb  = (float*)alloc((size_t)32 * 129 * 128 * 4);
  float* attn = (float*)alloc((size_t)32 * 128 * 128 * 4);
  float* ybuf = (float*)alloc((size_t)32 * 128 * 129 * 4);
  float* part = (float*)alloc((size_t)32 * 4 * 1024 * 4);
  float* feat = (float*)alloc((size_t)32 * 1024 * 4);
  float* h1   = (float*)alloc((size_t)32 * 256 * 4);
  _Float16* t1w1 = (_Float16*)alloc((size_t)64 * 32 * 2);
  _Float16* t1w2 = (_Float16*)alloc((size_t)64 * 64 * 2);
  _Float16* t1w3 = (_Float16*)alloc((size_t)128 * 64 * 2);
  _Float16* t2w1 = (_Float16*)alloc((size_t)128 * 160 * 2);
  _Float16* t2w2 = (_Float16*)alloc((size_t)128 * 128 * 2);
  _Float16* t2w3 = (_Float16*)alloc((size_t)256 * 128 * 2);
  _Float16* t3w1 = (_Float16*)alloc((size_t)256 * 288 * 2);
  _Float16* t3w2 = (_Float16*)alloc((size_t)512 * 256 * 2);
  _Float16* t3w3 = (_Float16*)alloc((size_t)1024 * 512 * 2);
  float* t1b1 = (float*)alloc(64 * 4);
  float* t1b2 = (float*)alloc(64 * 4);
  float* t1b3 = (float*)alloc(128 * 4);
  float* t2b1 = (float*)alloc(128 * 4);
  float* t2b2 = (float*)alloc(128 * 4);
  float* t2b3 = (float*)alloc(256 * 4);
  float* t3b1 = (float*)alloc(256 * 4);
  float* t3b2 = (float*)alloc(512 * 4);
  float* t3b3 = (float*)alloc(1024 * 4);

  // ---- fold affine into fragment-packed f16 weights ----
  auto prep = [&](int base, int blkIdx, _Float16* Wo, float* bo,
                  int cin, int cinP, int cout) {
    int total = cout * cinP;
    k_prep_w<<<(total + 255) / 256, 256, 0, stream>>>(
        F(base + 4 * blkIdx + 0), F(base + 4 * blkIdx + 1),
        F(base + 4 * blkIdx + 2), F(base + 4 * blkIdx + 3),
        Wo, bo, cin, cinP, cout);
  };
  prep(iR1, 0, t1w1, t1b1, 6, 32, 64);
  prep(iR1, 1, t1w2, t1b2, 64, 64, 64);
  prep(iR1, 2, t1w3, t1b3, 64, 64, 128);
  prep(iR2, 0, t2w1, t2b1, 131, 160, 128);
  prep(iR2, 1, t2w2, t2b2, 128, 128, 128);
  prep(iR2, 2, t2w3, t2b3, 128, 128, 256);
  prep(iR3, 0, t3w1, t3b1, 259, 288, 256);
  prep(iR3, 1, t3w2, t3b2, 256, 256, 512);
  prep(iR3, 2, t3w3, t3b3, 512, 512, 1024);

  // ---- pipeline ----
  k_tower1<<<32 * 512, 128, 0, stream>>>(xt, yt, t1w1, t1w2, t1w3,
                                         t1b1, t1b2, t1b3, x1);
  k_tower2<<<32 * 128, 128, 0, stream>>>(x1, t2w1, t2w2, t2w3,
                                         t2b1, t2b2, t2b3, x2);
  {
    int n = 3 * 32 * 128 * 129;
    k_nl_proj<<<(n + 255) / 256, 256, 0, stream>>>(
        x2, F(iNL + 0), F(iNL + 1), F(iNL + 2), F(iNL + 3),
        F(iNL + 4), F(iNL + 5), gx, thb, phb);
  }
  k_nl_attn<<<32 * 128, 128, 0, stream>>>(thb, phb, attn);
  {
    int n = 32 * 128 * 129;
    k_nl_y<<<(n + 255) / 256, 256, 0, stream>>>(attn, gx, ybuf);
  }
  {
    int n = 32 * 259 * 128;
    k_nl_out<<<(n + 255) / 256, 256, 0, stream>>>(ybuf, F(iNL + 6), F(iNL + 7), x2);
  }
  k_tower3<<<32 * 4, 128, 0, stream>>>(x2, t3w1, t3w2, t3w3,
                                       t3b1, t3b2, t3b3, part);
  k_final_max_bn<<<(32 * 1024 + 255) / 256, 256, 0, stream>>>(part, F(46), F(47), feat);
  k_fc1<<<(32 * 256 + 255) / 256, 256, 0, stream>>>(feat, F(48), F(49), F(50), F(51), h1);
  k_fc2<<<(32 * 60 + 255) / 256, 256, 0, stream>>>(h1, F(52), F(53), (float*)d_out);
}
